// VITONGMM_52501680226679
// MI455X (gfx1250) — matrix operand order repaired and checked by reading, not compiled, not run on hardware
//
#include <hip/hip_runtime.h>
#include <hip/hip_bf16.h>

// ---------------------------------------------------------------------------
// VITON-GMM forward for MI455X (gfx1250).
//  * convs = implicit GEMM on v_wmma_f32_16x16x32_f16, M=64 per block
//    (4 WMMAs per K-chunk sharing one im2col B fragment)
//  * weights prepacked to f16 K-major (32-padded) + K-index lookup table,
//    so the inner loop has no divisions and A-staging is a raw 16B copy
//    routed through global_load_async_to_lds_b128 (ASYNCcnt)
//  * training-mode BN batch statistics via shfl_xor pre-reduction + f32
//    global atomics; fused BN+ReLU+f16 conversion
//  * correlation / linear / NaN-scrub + bilinear upsample tail kernels
// ---------------------------------------------------------------------------

typedef __attribute__((ext_vector_type(16))) _Float16 v16h;
typedef __attribute__((ext_vector_type(4)))  _Float16 v4h;
typedef __attribute__((ext_vector_type(8)))  float    v8f;
typedef __attribute__((ext_vector_type(4)))  int      i32x4;
typedef _Float16 f16;

#define NBATCH 16
#define AS1 __attribute__((address_space(1)))
#define AS3 __attribute__((address_space(3)))

// ---- global -> LDS 16-byte copy, async-to-LDS when the toolchain has it ----
__device__ __forceinline__ void g2l_copy16(f16* lds, const f16* g) {
#if __has_builtin(__builtin_amdgcn_global_load_async_to_lds_b128)
    __builtin_amdgcn_global_load_async_to_lds_b128(
        (AS1 i32x4*)g, (AS3 i32x4*)lds, 0, 0);
#else
    *(uint4*)lds = *(const uint4*)g;
#endif
}
__device__ __forceinline__ void wait_async_copies() {
#if __has_builtin(__builtin_amdgcn_global_load_async_to_lds_b128)
#if __has_builtin(__builtin_amdgcn_s_wait_asynccnt)
    __builtin_amdgcn_s_wait_asynccnt(0);
#else
    asm volatile("s_wait_asynccnt 0" ::: "memory");
#endif
#endif
}

// ---------------------------------------------------------------------------
// Per-layer weight prepack: f32 [cout,cinW,k,k] -> f16 [cout,Kpad] K-major
// (zero padded), plus K-index table ktab[kk] = {ci*Hin*Win, ky, kx, 0}.
// ---------------------------------------------------------------------------
__global__ __launch_bounds__(256) void prepack_kernel(
        const float* __restrict__ w, f16* __restrict__ wp, int4* __restrict__ ktab,
        int cout, int cinW, int k, int Keff, int Kpad, int HinWin) {
    const int kk2 = k * k;
    const int total = cout * Kpad;
    for (int i = blockIdx.x * blockDim.x + threadIdx.x; i < total + Kpad;
         i += gridDim.x * blockDim.x) {
        if (i < total) {
            int co = i / Kpad, kk = i - co * Kpad;
            float v = 0.0f;
            if (kk < Keff) {
                int ci = kk / kk2, r = kk - ci * kk2;
                int ky = r / k, kx = r - ky * k;
                v = w[((co * cinW + ci) * k + ky) * k + kx];
            }
            wp[i] = (f16)v;
        } else {
            int kk = i - total;
            int4 e = {0, 0, 0, 0};
            if (kk < Keff) {
                int ci = kk / kk2, r = kk - ci * kk2;
                int ky = r / k, kx = r - ky * k;
                e.x = ci * HinWin; e.y = ky; e.z = kx;
            }
            ktab[kk] = e;
        }
    }
}

// ---------------------------------------------------------------------------
// Implicit-GEMM conv.  block = 4 waves; blockIdx.x -> 64-channel cout tile,
// blockIdx.y*4 + wave -> 16-pixel tile.  K in chunks of 32 f16.
// Fragments use contiguous k = 16*(lane/16) + e in both A and B (wave32).
// ---------------------------------------------------------------------------
struct ConvArgs {
    const f16*  in;        // NCHW f16, cinPhys channels
    const f16*  wp;        // prepacked weights [cout, Kpad]
    const int4* ktab;      // [Kpad]
    const float* bias;
    float* raw;            // NCHW f32 conv output
    float* stats;          // [0..cout)=sum, [cout..2cout)=sumsq
    int cinPhys, Hin, Win;
    int cout, stride, pad, Hout, Wout, Kpad;
};

__global__ __launch_bounds__(128) void conv_wmma(ConvArgs a) {
    const int lane = threadIdx.x & 31;
    const int wv   = threadIdx.x >> 5;
    const int ct   = blockIdx.x * 64;                  // cout tile base
    const int HW   = a.Hout * a.Wout;
    const int pixBase = (blockIdx.y * 4 + wv) * 16;

    __shared__ __align__(16) f16 aLds[64 * 32];        // 4 KB, block-shared
    __shared__ __align__(16) f16 bLds[4][32 * 16];     // 1 KB per wave

    // hoist per-pixel decomposition out of the K loop (uniform per wave)
    int pOff[16], pOy[16], pOx[16];
    const int HinWin = a.Hin * a.Win;
    #pragma unroll
    for (int p = 0; p < 16; ++p) {
        int pix = pixBase + p;
        int nB  = pix / HW;
        int hw  = pix - nB * HW;
        int oy  = hw / a.Wout;
        int ox  = hw - oy * a.Wout;
        pOff[p] = nB * a.cinPhys * HinWin;
        pOy[p]  = oy * a.stride - a.pad;
        pOx[p]  = ox * a.stride - a.pad;
    }

    v8f acc[4] = {{}, {}, {}, {}};
    const int Kch = a.Kpad >> 5;

    for (int kb = 0; kb < Kch; ++kb) {
        const int kbase = kb * 32;
        __syncthreads();

        // --- A tile: 64 rows x 32 halves = 4KB; each thread copies 32B ---
        {
            int r = threadIdx.x >> 1, seg = (threadIdx.x & 1) * 16;
            const f16* src = a.wp + (size_t)(ct + r) * a.Kpad + kbase + seg;
            g2l_copy16(&aLds[r * 32 + seg],     src);
            g2l_copy16(&aLds[r * 32 + seg + 8], src + 8);
        }

        // --- B tile (per wave): lane = K row, im2col gather of 16 pixels ---
        {
            int4 kt = a.ktab[kbase + lane];   // {ci*HinWin, ky, kx} (0 if pad)
            #pragma unroll
            for (int p = 0; p < 16; ++p) {
                int iy = pOy[p] + kt.y;
                int ix = pOx[p] + kt.z;
                f16 v = (f16)0.0f;
                if ((unsigned)iy < (unsigned)a.Hin && (unsigned)ix < (unsigned)a.Win)
                    v = a.in[pOff[p] + kt.x + iy * a.Win + ix];
                bLds[wv][lane * 16 + p] = v;
            }
        }
        wait_async_copies();
        __syncthreads();

        // --- fragments per wave32 WMMA layout ---
        v16h af[4], bf;
        const int mrow = lane & 15;
        const int hiA  = (lane >= 16) ? 8  : 0;
        const int hiB  = (lane >= 16) ? 16 : 0;
        #pragma unroll
        for (int e = 0; e < 16; ++e) {
            int kkA = hiB + e;
            #pragma unroll
            for (int t = 0; t < 4; ++t)
                af[t][e] = aLds[(t * 16 + mrow) * 32 + kkA];
            bf[e] = bLds[wv][(hiB + e) * 16 + mrow];
        }
        #pragma unroll
        for (int t = 0; t < 4; ++t)
            acc[t] = __builtin_amdgcn_wmma_f32_16x16x32_f16(
                         false, af[t], false, bf, (short)0, acc[t], false, false);
    }

    // --- epilogue: +bias, raw f32 store, BN statistics ---
    const int n   = lane & 15;
    const int pix = pixBase + n;
    const int nB  = pix / HW;
    const int hw  = pix - nB * HW;
    const int rbase = (lane >= 16) ? 8 : 0;

    float sv[32], qv[32];
    #pragma unroll
    for (int t = 0; t < 4; ++t) {
        #pragma unroll
        for (int r = 0; r < 8; ++r) {
            int cch = ct + t * 16 + rbase + r;
            float v = acc[t][r] + a.bias[cch];
            a.raw[((size_t)(nB * a.cout + cch)) * HW + hw] = v;
            sv[t * 8 + r] = v;
            qv[t * 8 + r] = v * v;
        }
    }
    #pragma unroll
    for (int off = 1; off < 16; off <<= 1) {            // masks < 16: stays in half
        #pragma unroll
        for (int r = 0; r < 32; ++r) {
            sv[r] += __shfl_xor(sv[r], off);
            qv[r] += __shfl_xor(qv[r], off);
        }
    }
    if ((lane & 15) == 0) {
        #pragma unroll
        for (int t = 0; t < 4; ++t) {
            #pragma unroll
            for (int r = 0; r < 8; ++r) {
                int cch = ct + t * 16 + rbase + r;
                atomicAdd(&a.stats[cch],          sv[t * 8 + r]);
                atomicAdd(&a.stats[a.cout + cch], qv[t * 8 + r]);
            }
        }
    }
}

// ---------------------------------------------------------------------------
// Fused training-mode BN (batch stats) + ReLU + f32->f16, vectorized x4.
// HW is a multiple of 4 for every layer.
// ---------------------------------------------------------------------------
__global__ __launch_bounds__(256) void bnrelu_kernel(
        const float4* __restrict__ raw, const float* __restrict__ stats,
        const float* __restrict__ gamma, const float* __restrict__ beta,
        v4h* __restrict__ act, int C, int HW4, float inv, int total4) {
    for (int i = blockIdx.x * blockDim.x + threadIdx.x; i < total4;
         i += gridDim.x * blockDim.x) {
        int c = (i / HW4) % C;
        float m  = stats[c] * inv;
        float vv = stats[C + c] * inv - m * m;
        float rs = __frsqrt_rn(vv + 1e-5f);
        float g = gamma[c], bt = beta[c];
        float4 x = raw[i];
        v4h o;
        o[0] = (f16)fmaxf(g * ((x.x - m) * rs) + bt, 0.0f);
        o[1] = (f16)fmaxf(g * ((x.y - m) * rs) + bt, 0.0f);
        o[2] = (f16)fmaxf(g * ((x.z - m) * rs) + bt, 0.0f);
        o[3] = (f16)fmaxf(g * ((x.w - m) * rs) + bt, 0.0f);
        act[i] = o;
    }
}

// ---------------------------------------------------------------------------
// Prep: cloth f32->f16 (zero parsing channels skipped); grayscale mean -> f16
// ---------------------------------------------------------------------------
__global__ __launch_bounds__(256) void prep_kernel(
        const float* __restrict__ cloth, const float* __restrict__ person,
        f16* __restrict__ cloth16, f16* __restrict__ gray16,
        int nCloth, int nGray) {
    const int stride = gridDim.x * blockDim.x;
    const int tid = blockIdx.x * blockDim.x + threadIdx.x;
    for (int j = tid; j < nCloth; j += stride) cloth16[j] = (f16)cloth[j];
    for (int j = tid; j < nGray; j += stride) {
        int b  = j >> 16;
        int hw = j & 65535;
        const float* p = person + (size_t)b * 3 * 65536 + hw;
        gray16[j] = (f16)((p[0] + p[65536] + p[131072]) * (1.0f / 3.0f));
    }
}

// ---------------------------------------------------------------------------
// Windowed-displacement correlation (md=5): 121 maps + repeat-to-192, f16 out
// ---------------------------------------------------------------------------
__global__ __launch_bounds__(128) void corr_kernel(
        const f16* __restrict__ fA, const f16* __restrict__ fB,
        f16* __restrict__ corr) {
    const int wv = threadIdx.x >> 5, lane = threadIdx.x & 31;
    const int q  = blockIdx.x * 4 + wv;
    const int b  = q >> 8;
    const int yx = q & 255;
    const int y  = yx >> 4, x = yx & 15;

    const f16* Ab = fA + (size_t)b * 512 * 256;
    const f16* Bb = fB + (size_t)b * 512 * 256;

    float av[16];
    #pragma unroll
    for (int t = 0; t < 16; ++t)
        av[t] = (float)Ab[(lane + 32 * t) * 256 + yx];

    for (int d = 0; d < 121; ++d) {
        int di = d / 11 - 5, dj = d % 11 - 5;
        int ys = y + di, xs = x + dj;
        float acc = 0.0f;
        if ((unsigned)ys < 16u && (unsigned)xs < 16u) {
            int o = ys * 16 + xs;
            #pragma unroll
            for (int t = 0; t < 16; ++t)
                acc += av[t] * (float)Bb[(lane + 32 * t) * 256 + o];
        }
        #pragma unroll
        for (int off = 1; off < 32; off <<= 1) acc += __shfl_xor(acc, off);
        if (lane == 0) {
            f16 h = (f16)acc;
            corr[((size_t)(b * 192 + d)) * 256 + yx] = h;
            if (d < 71)
                corr[((size_t)(b * 192 + d + 121)) * 256 + yx] = h;
        }
    }
}

// ---------------------------------------------------------------------------
// Linear: [16,1024] @ W^T + b  (x row staged in LDS)
// ---------------------------------------------------------------------------
__global__ __launch_bounds__(256) void linear_kernel(
        const f16* __restrict__ xin, const float* __restrict__ W,
        const float* __restrict__ bvec, float* __restrict__ out) {
    __shared__ float xs[1024];
    const int b = blockIdx.x, t = threadIdx.x;
    for (int i = t; i < 1024; i += 256) xs[i] = (float)xin[b * 1024 + i];
    __syncthreads();
    for (int o = t; o < 512; o += 256) {
        const float* wr = W + (size_t)o * 1024;
        float acc = bvec[o];
        #pragma unroll 8
        for (int kk = 0; kk < 1024; ++kk) acc += wr[kk] * xs[kk];
        out[b * 512 + o] = acc;
    }
}

// ---------------------------------------------------------------------------
// NaN/Inf scrub + clip(+-100) + x16 scale + bilinear align_corners 16 -> 256
// ---------------------------------------------------------------------------
__device__ __forceinline__ float fix_flow(float v) {
    if (!(fabsf(v) <= 3.4e38f)) v = 0.0f;          // NaN / +-Inf -> 0
    v = fminf(fmaxf(v, -100.0f), 100.0f);
    return v * 16.0f;                               // W/Wc == H/Hc == 16
}

__global__ __launch_bounds__(256) void upsample_kernel(
        const float* __restrict__ lin, float* __restrict__ out) {
    int i = blockIdx.x * blockDim.x + threadIdx.x;  // 16*2*256*256 total
    if (i >= 16 * 2 * 256 * 256) return;
    int ox = i & 255;
    int t  = i >> 8;
    int oy = t & 255; t >>= 8;
    int ch = t & 1;
    int b  = t >> 1;
    float fy = oy * (15.0f / 255.0f);
    float fx = ox * (15.0f / 255.0f);
    int y0 = (int)fy; int y1 = (y0 + 1 < 15) ? y0 + 1 : 15;
    int x0 = (int)fx; int x1 = (x0 + 1 < 15) ? x0 + 1 : 15;
    float wy = fy - (float)y0, wx = fx - (float)x0;
    const float* F = lin + b * 512 + ch * 256;
    float v00 = fix_flow(F[y0 * 16 + x0]);
    float v01 = fix_flow(F[y0 * 16 + x1]);
    float v10 = fix_flow(F[y1 * 16 + x0]);
    float v11 = fix_flow(F[y1 * 16 + x1]);
    float top = v00 * (1.0f - wx) + v01 * wx;
    float bot = v10 * (1.0f - wx) + v11 * wx;
    out[i] = top * (1.0f - wy) + bot * wy;
}

// ---------------------------------------------------------------------------
extern "C" void kernel_launch(void* const* d_in, const int* in_sizes, int n_in,
                              void* d_out, int out_size, void* d_ws, size_t ws_size,
                              hipStream_t stream) {
    (void)in_sizes; (void)n_in; (void)out_size; (void)ws_size;

    const float* cloth  = (const float*)d_in[0];
    const float* person = (const float*)d_in[1];
    auto P = [&](int base, int layer, int which) {
        return (const float*)d_in[base + 4 * layer + which];
    };
    const float* lin_w = (const float*)d_in[58];
    const float* lin_b = (const float*)d_in[59];

    // ---- workspace bump allocator (256B aligned) ----
    char*  ws  = (char*)d_ws;
    size_t off = 0;
    auto alloc = [&](size_t bytes) -> void* {
        void* p = ws + off;
        off += (bytes + 255) & ~(size_t)255;
        return p;
    };
    const size_t maxActElems = (size_t)NBATCH * 64 * 128 * 128;
    f16*   actA   = (f16*)  alloc(maxActElems * sizeof(f16));
    f16*   actB   = (f16*)  alloc(maxActElems * sizeof(f16));
    float* raw    = (float*)alloc(maxActElems * sizeof(float));
    float* stats  = (float*)alloc(1024 * sizeof(float));
    f16*   gray   = (f16*)  alloc((size_t)NBATCH * 65536 * sizeof(f16));
    f16*   featA  = (f16*)  alloc((size_t)NBATCH * 512 * 256 * sizeof(f16));
    f16*   corrb  = (f16*)  alloc((size_t)NBATCH * 192 * 256 * sizeof(f16));
    float* linout = (float*)alloc((size_t)NBATCH * 512 * sizeof(float));

    // ---- layer table ----
    struct LD {
        int base, layer;               // d_in param location
        int cinW, cinEff, cinPhys;     // weight-cin, effective-cin, physical-cin
        int cout, k, s, p, Hin, Win;
        f16* wp; int4* ktab; int Kpad; // filled below
    };
    LD L[14] = {
        // stack A (zero parsing -> effective Cin 3 of 22)
        { 2,0, 22,  3,  3,  64,4,2,1, 256,256, 0,0,0},
        { 2,1, 64, 64, 64, 128,4,2,1, 128,128, 0,0,0},
        { 2,2,128,128,128, 256,4,2,1,  64, 64, 0,0,0},
        { 2,3,256,256,256, 512,4,2,1,  32, 32, 0,0,0},
        { 2,4,512,512,512, 512,3,1,1,  16, 16, 0,0,0},
        // stack B
        {22,0,  1,  1,  1,  64,4,2,1, 256,256, 0,0,0},
        {22,1, 64, 64, 64, 128,4,2,1, 128,128, 0,0,0},
        {22,2,128,128,128, 256,4,2,1,  64, 64, 0,0,0},
        {22,3,256,256,256, 512,4,2,1,  32, 32, 0,0,0},
        {22,4,512,512,512, 512,3,1,1,  16, 16, 0,0,0},
        // stack R
        {42,0,192,192,192, 512,4,2,1,  16, 16, 0,0,0},
        {42,1,512,512,512, 256,4,2,1,   8,  8, 0,0,0},
        {42,2,256,256,256, 128,3,1,1,   4,  4, 0,0,0},
        {42,3,128,128,128,  64,3,1,1,   4,  4, 0,0,0},
    };

    // ---- prepack all weights (f16 K-major) + K-index tables ----
    for (int i = 0; i < 14; ++i) {
        int Keff = L[i].cinEff * L[i].k * L[i].k;
        L[i].Kpad = (Keff + 31) & ~31;
        L[i].wp   = (f16*) alloc((size_t)L[i].cout * L[i].Kpad * sizeof(f16));
        L[i].ktab = (int4*)alloc((size_t)L[i].Kpad * sizeof(int4));
        int total = L[i].cout * L[i].Kpad + L[i].Kpad;
        int blocks = (total + 255) / 256; if (blocks > 2048) blocks = 2048;
        prepack_kernel<<<blocks, 256, 0, stream>>>(
            P(L[i].base, L[i].layer, 0), L[i].wp, L[i].ktab,
            L[i].cout, L[i].cinW, L[i].k, Keff, L[i].Kpad,
            L[i].Hin * L[i].Win);
    }

    int Ho = 0, Wo = 0;
    auto conv_bn = [&](int li, const f16* in, f16* actout) {
        const LD& l = L[li];
        Ho = (l.Hin + 2 * l.p - l.k) / l.s + 1;
        Wo = (l.Win + 2 * l.p - l.k) / l.s + 1;
        const int HW = Ho * Wo;
        const int Npix = NBATCH * HW;                 // multiple of 64 everywhere
        (void)hipMemsetAsync(stats, 0, 2 * l.cout * sizeof(float), stream);
        ConvArgs ca;
        ca.in = in; ca.wp = l.wp; ca.ktab = l.ktab;
        ca.bias = P(l.base, l.layer, 1);
        ca.raw = raw; ca.stats = stats;
        ca.cinPhys = l.cinPhys; ca.Hin = l.Hin; ca.Win = l.Win;
        ca.cout = l.cout; ca.stride = l.s; ca.pad = l.p;
        ca.Hout = Ho; ca.Wout = Wo; ca.Kpad = l.Kpad;
        dim3 grid(l.cout / 64, Npix / 64);
        conv_wmma<<<grid, 128, 0, stream>>>(ca);
        const int total4 = (NBATCH * l.cout * HW) / 4;
        int blocks = (total4 + 255) / 256; if (blocks > 8192) blocks = 8192;
        bnrelu_kernel<<<blocks, 256, 0, stream>>>(
            (const float4*)raw, stats, P(l.base, l.layer, 2), P(l.base, l.layer, 3),
            (v4h*)actout, l.cout, HW / 4, 1.0f / (float)(NBATCH * HW), total4);
    };

    // ---- prep ----
    prep_kernel<<<4096, 256, 0, stream>>>(cloth, person, actA, gray,
                                          NBATCH * 3 * 65536, NBATCH * 65536);
    // ---- stack A ----
    conv_bn(0, actA, actB);
    conv_bn(1, actB, actA);
    conv_bn(2, actA, actB);
    conv_bn(3, actB, actA);
    conv_bn(4, actA, featA);
    // ---- stack B ----
    conv_bn(5, gray, actA);
    conv_bn(6, actA, actB);
    conv_bn(7, actB, actA);
    conv_bn(8, actA, actB);
    conv_bn(9, actB, actA);                          // featB in actA
    // ---- correlation ----
    corr_kernel<<<(NBATCH * 256) / 4, 128, 0, stream>>>(featA, actA, corrb);
    // ---- stack R ----
    conv_bn(10, corrb, actB);
    conv_bn(11, actB, actA);
    conv_bn(12, actA, actB);
    conv_bn(13, actB, actA);
    // ---- linear + flow postprocess + upsample ----
    linear_kernel<<<NBATCH, 256, 0, stream>>>(actA, lin_w, lin_b, linout);
    upsample_kernel<<<(NBATCH * 2 * 256 * 256) / 256, 256, 0, stream>>>(
        linout, (float*)d_out);
}
